// VAECW_88682484728322
// MI455X (gfx1250) — matrix operand order repaired and checked by reading, not compiled, not run on hardware
//
#include <hip/hip_runtime.h>
#include <hip/hip_bf16.h>

// ---------------- problem constants ----------------
#define BATCH     128
#define DIMC      256
#define EMBED     16
#define BOOK      512
#define CWDIM     4096      // DIMC*EMBED
#define ZD        512
#define HENC      2048
#define HDEC      2048
#define NPSEUDO   400
#define MROWS     528       // BATCH + NPSEUDO
#define MPAD      640       // 528 padded to multiple of 128

// output offsets (floats, in tuple return order)
#define OFF_MU    0
#define OFF_LV    65536
#define OFF_PMU   131072
#define OFF_PLV   335872
#define OFF_Z     540672
#define OFF_CW    606208
#define OFF_DIST  1130496
#define OFF_IDX   17907712

// workspace offsets (floats)
#define WS_XALL   0          // 640*4096
#define WS_H      2621440    // 640*2048
#define WS_ENC    3932160    // 640*1024
#define WS_Z      4587520    // 128*512
#define WS_H2     4653056    // 128*2048

#ifndef __has_builtin
#define __has_builtin(x) 0
#endif
#if __has_builtin(__builtin_amdgcn_global_load_async_to_lds_b128) && \
    __has_builtin(__builtin_amdgcn_s_wait_asynccnt)
#define USE_ASYNC_LDS 1
#else
#define USE_ASYNC_LDS 0
#endif

typedef __bf16 v16bf __attribute__((ext_vector_type(16)));
typedef float  v8f   __attribute__((ext_vector_type(8)));
typedef int    v4i   __attribute__((ext_vector_type(4)));

#define AS1 __attribute__((address_space(1)))
#define AS3 __attribute__((address_space(3)))

union Frag { unsigned int u[8]; v16bf v; };

// round-to-nearest-even fp32 -> bf16, packed pair into one dword
__device__ __forceinline__ unsigned int pack_bf16(float a, float b) {
    unsigned int ua = __float_as_uint(a);
    unsigned int ub = __float_as_uint(b);
    unsigned int ra = (ua + 0x7FFFu + ((ua >> 16) & 1u)) >> 16;
    unsigned int rb = (ub + 0x7FFFu + ((ub >> 16) & 1u)) >> 16;
    return (rb << 16) | (ra & 0xFFFFu);
}

// ---------------- build permuted encoder input ----------------
// xall[row, e*256+d] = x[row, d*16+e] for row<128 ; pseudo rows already [E,D] flat ; pad rows = 0
__global__ __launch_bounds__(256) void pack_xall(const float* __restrict__ x,
                                                 const float* __restrict__ pseudo,
                                                 float* __restrict__ xall) {
    int i   = blockIdx.x * 256 + threadIdx.x;   // 0 .. 640*4096-1
    int row = i >> 12;
    int col = i & 4095;
    float v;
    if (row < BATCH) {
        int e  = col >> 8;       // embed index
        int dc = col & 255;      // code-dim index
        v = x[(size_t)row * CWDIM + dc * EMBED + e];
    } else if (row < MROWS) {
        v = pseudo[(size_t)(row - BATCH) * CWDIM + col];
    } else {
        v = 0.0f;
    }
    xall[i] = v;
}

// ---------------- generic bf16-WMMA GEMM: C = [relu](A @ W + bias) ----------------
// A: [gridDim.y*128, K] fp32 (padded rows valid), W: [K,N] fp32, C: [gridDim.y*128, N] fp32
// 256 threads = 8 waves; WG tile 128x128, BK=32; wave tile 32x64 (2x4 WMMA accums)
// W tile is staged raw (fp32) into LDS via GLOBAL_LOAD_ASYNC_TO_LDS_B128 when available,
// so the async engine streams weights while WMMAs execute; converted at gather time.
template <int RELU>
__global__ __launch_bounds__(256) void gemm_bf16(const float* __restrict__ A,
                                                 const float* __restrict__ W,
                                                 const float* __restrict__ bias,
                                                 float* __restrict__ C,
                                                 int N, int K) {
    __shared__ unsigned int ldsA[128 * 17];          // [m][k/2] packed bf16 pairs
#if USE_ASYNC_LDS
    __shared__ __align__(16) float ldsBf[32 * 128];  // raw fp32 W tile [k][n]
#else
    __shared__ unsigned int ldsB[128 * 17];          // [n][k/2] packed bf16 pairs
#endif
    const int t     = threadIdx.x;
    const int lane  = t & 31;
    const int wid   = t >> 5;
    const int m0    = blockIdx.y * 128;
    const int n0    = blockIdx.x * 128;
    const int waveM = (wid >> 1) * 32;
    const int waveN = (wid & 1) * 64;
    const int lr    = lane & 15;
    const int lh    = lane >> 4;

    v8f acc[2][4];
    #pragma unroll
    for (int i = 0; i < 2; ++i)
        #pragma unroll
        for (int j = 0; j < 4; ++j)
            acc[i][j] = v8f{};

    for (int k0 = 0; k0 < K; k0 += 32) {
        __syncthreads();
#if USE_ASYNC_LDS
        // async-copy W tile 32x128 fp32 (16KB) straight into LDS, 16B per lane
        #pragma unroll
        for (int i = 0; i < 4; ++i) {
            int e  = i * 256 + t;     // 0..1023 chunks of 4 floats
            int ki = e >> 5;          // k row 0..31
            int nq = e & 31;          // 16B chunk within row
            __builtin_amdgcn_global_load_async_to_lds_b128(
                (AS1 v4i*)(W + (size_t)(k0 + ki) * N + n0 + nq * 4),
                (AS3 v4i*)(ldsBf + ki * 128 + nq * 4),
                0, 0);
        }
#else
        // stage W tile 32x128 transposed into [n][k/2] packed bf16
        #pragma unroll
        for (int i = 0; i < 8; ++i) {
            int e  = i * 256 + t;
            int ni = e & 127;
            int kp = e >> 7;
            size_t r = (size_t)(k0 + kp * 2) * N + n0 + ni;
            ldsB[ni * 17 + kp] = pack_bf16(W[r], W[r + N]);
        }
#endif
        // stage A tile 128x32 (fp32 -> bf16 pairs): 2048 dwords (overlaps with async copy)
        #pragma unroll
        for (int i = 0; i < 8; ++i) {
            int e  = i * 256 + t;
            int mi = e >> 4;
            int kp = e & 15;
            const float2 f2 = *(const float2*)(A + (size_t)(m0 + mi) * K + k0 + kp * 2);
            ldsA[mi * 17 + kp] = pack_bf16(f2.x, f2.y);
        }
#if USE_ASYNC_LDS
        __builtin_amdgcn_s_wait_asynccnt(0);
#endif
        __syncthreads();

        // gather fragments in contiguous k order (k = 16*lane-half + element), same for A and B
        Frag af[2], bfr[4];
        #pragma unroll
        for (int ti = 0; ti < 2; ++ti) {
            int m = waveM + ti * 16 + lr;
            #pragma unroll
            for (int v = 0; v < 8; ++v) {
                int kp = v + 8 * lh;                      // A: K = lane-half*16 + 2v
                af[ti].u[v] = ldsA[m * 17 + kp];
            }
        }
        #pragma unroll
        for (int tj = 0; tj < 4; ++tj) {
            int n = waveN + tj * 16 + lr;
            #pragma unroll
            for (int v = 0; v < 8; ++v) {
#if USE_ASYNC_LDS
                int k = 16 * lh + 2 * v;                  // B: K = lane-half*16 + 2v
                bfr[tj].u[v] = pack_bf16(ldsBf[k * 128 + n], ldsBf[(k + 1) * 128 + n]);
#else
                int kp = v + 8 * lh;
                bfr[tj].u[v] = ldsB[n * 17 + kp];
#endif
            }
        }
        #pragma unroll
        for (int ti = 0; ti < 2; ++ti)
            #pragma unroll
            for (int tj = 0; tj < 4; ++tj)
                acc[ti][tj] = __builtin_amdgcn_wmma_f32_16x16x32_bf16(
                    false, af[ti].v, false, bfr[tj].v, (short)0, acc[ti][tj], false, false);
    }

    // writeback, guard-free (grid exactly tiles the padded matrices)
    // C/D layout: lane%16 = N, vgpr i -> M = i + 8*(lane/16)
    #pragma unroll
    for (int ti = 0; ti < 2; ++ti) {
        #pragma unroll
        for (int tj = 0; tj < 4; ++tj) {
            int col  = n0 + waveN + tj * 16 + lr;
            float bv = bias[col];
            #pragma unroll
            for (int i = 0; i < 8; ++i) {
                int row = m0 + waveM + ti * 16 + lh * 8 + i;
                float v = acc[ti][tj][i] + bv;
                if (RELU) v = fmaxf(v, 0.0f);
                C[(size_t)row * N + col] = v;
            }
        }
    }
}

// ---------------- split enc -> mu/log_var/pseudo_* and reparameterize ----------------
__global__ __launch_bounds__(256) void reparam_kernel(const float* __restrict__ enc,
                                                      const float* __restrict__ eps,
                                                      float* __restrict__ out,
                                                      float* __restrict__ z_ws) {
    int i = blockIdx.x * 256 + threadIdx.x;   // 0 .. 528*512-1
    if (i >= MROWS * ZD) return;
    int r = i >> 9;
    int j = i & 511;
    float mu = enc[(size_t)r * (2 * ZD) + j];
    float lv = enc[(size_t)r * (2 * ZD) + ZD + j];
    if (r < BATCH) {
        float z = mu + eps[(size_t)r * ZD + j] * __expf(0.5f * lv);
        out[OFF_MU + r * ZD + j] = mu;
        out[OFF_LV + r * ZD + j] = lv;
        out[OFF_Z  + r * ZD + j] = z;
        z_ws[r * ZD + j] = z;
    } else {
        int p = r - BATCH;
        out[OFF_PMU + p * ZD + j] = mu;
        out[OFF_PLV + p * ZD + j] = lv;
    }
}

// ---------------- codebook distances: one WG per (d, 128-code chunk) ----------------
// dist[b,d,k] = ||xr[b,d,:]||^2 - 2*<xr,book[d,k,:]> + ||book[d,k,:]||^2
__global__ __launch_bounds__(256) void dist_kernel(const float* __restrict__ cw,   // [128,4096]
                                                   const float* __restrict__ book, // [256,512,16]
                                                   float* __restrict__ dist) {     // [128,256,512]
    __shared__ unsigned int ldsA[128 * 17];
    __shared__ unsigned int ldsB[128 * 17];
    __shared__ float x2s[128];
    __shared__ float b2s[128];
    const int t     = threadIdx.x;
    const int lane  = t & 31;
    const int wid   = t >> 5;
    const int d     = blockIdx.y;
    const int n0    = blockIdx.x * 128;
    const int waveM = (wid >> 1) * 32;
    const int waveN = (wid & 1) * 64;
    const int lr    = lane & 15;
    const int lh    = lane >> 4;

    // stage A (xr rows, K padded 16->32 with zeros) and B (codebook rows) in one sweep
    #pragma unroll
    for (int i = 0; i < 4; ++i) {
        int e  = i * 256 + t;            // 0..1023
        int mi = e >> 3;
        int kp = e & 7;
        const float2 a2 = *(const float2*)(cw + (size_t)mi * CWDIM + d * EMBED + kp * 2);
        ldsA[mi * 17 + kp]     = pack_bf16(a2.x, a2.y);
        ldsA[mi * 17 + 8 + kp] = 0u;
        const float2 b2 = *(const float2*)(book + ((size_t)d * BOOK + n0 + mi) * EMBED + kp * 2);
        ldsB[mi * 17 + kp]     = pack_bf16(b2.x, b2.y);
        ldsB[mi * 17 + 8 + kp] = 0u;
    }
    // fp32 squared norms (wave-uniform branch: waves 0-3 vs 4-7)
    if (t < 128) {
        float s = 0.0f;
        const float* src = cw + (size_t)t * CWDIM + d * EMBED;
        #pragma unroll
        for (int e = 0; e < EMBED; ++e) { float v = src[e]; s += v * v; }
        x2s[t] = s;
    } else {
        int n = t - 128;
        float s = 0.0f;
        const float* src = book + ((size_t)d * BOOK + n0 + n) * EMBED;
        #pragma unroll
        for (int e = 0; e < EMBED; ++e) { float v = src[e]; s += v * v; }
        b2s[n] = s;
    }
    __syncthreads();

    Frag af[2], bfr[4];
    #pragma unroll
    for (int ti = 0; ti < 2; ++ti) {
        int m = waveM + ti * 16 + lr;
        #pragma unroll
        for (int v = 0; v < 8; ++v) {
            int kp = v + 8 * lh;
            af[ti].u[v] = ldsA[m * 17 + kp];
        }
    }
    #pragma unroll
    for (int tj = 0; tj < 4; ++tj) {
        int n = waveN + tj * 16 + lr;
        #pragma unroll
        for (int v = 0; v < 8; ++v) {
            int kp = v + 8 * lh;
            bfr[tj].u[v] = ldsB[n * 17 + kp];
        }
    }
    v8f acc[2][4];
    #pragma unroll
    for (int ti = 0; ti < 2; ++ti)
        #pragma unroll
        for (int tj = 0; tj < 4; ++tj) {
            acc[ti][tj] = v8f{};
            acc[ti][tj] = __builtin_amdgcn_wmma_f32_16x16x32_bf16(
                false, af[ti].v, false, bfr[tj].v, (short)0, acc[ti][tj], false, false);
        }

    #pragma unroll
    for (int ti = 0; ti < 2; ++ti) {
        #pragma unroll
        for (int tj = 0; tj < 4; ++tj) {
            int cl    = waveN + tj * 16 + lr;    // code index within chunk
            float b2v = b2s[cl];
            #pragma unroll
            for (int i = 0; i < 8; ++i) {
                int row = waveM + ti * 16 + lh * 8 + i;
                float v = x2s[row] - 2.0f * acc[ti][tj][i] + b2v;
                dist[(size_t)row * (DIMC * BOOK) + (size_t)d * BOOK + n0 + cl] = v;
            }
        }
    }
}

// ---------------- argmin over K=512 per (b,d) row, wave-per-row ----------------
// cw_dist (67MB) fits in the 192MB L2, so this re-read is an L2 hit, not HBM traffic.
__global__ __launch_bounds__(256) void argmin_kernel(const float* __restrict__ dist,
                                                     float* __restrict__ idx_out) {
    const int lane = threadIdx.x & 31;
    const int wid  = threadIdx.x >> 5;
    const int row  = blockIdx.x * 8 + wid;           // 0 .. 32767
    const float* src = dist + (size_t)row * BOOK;
    float best = 3.402823466e38f;
    int bi = 0;
    #pragma unroll
    for (int k = lane; k < BOOK; k += 32) {
        float v = src[k];
        if (v < best) { best = v; bi = k; }          // lane-local ks increase -> first min kept
    }
    #pragma unroll
    for (int off = 16; off >= 1; off >>= 1) {
        float ov = __shfl_xor(best, off, 32);
        int   oi = __shfl_xor(bi, off, 32);
        if (ov < best || (ov == best && oi < bi)) { best = ov; bi = oi; }
    }
    if (lane == 0) idx_out[row] = (float)bi;
}

// ---------------- launch ----------------
extern "C" void kernel_launch(void* const* d_in, const int* in_sizes, int n_in,
                              void* d_out, int out_size, void* d_ws, size_t ws_size,
                              hipStream_t stream) {
    (void)in_sizes; (void)n_in; (void)out_size; (void)ws_size;
    const float* x        = (const float*)d_in[0];
    const float* codebook = (const float*)d_in[1];
    const float* pseudo   = (const float*)d_in[2];
    const float* eps      = (const float*)d_in[3];
    const float* W_e1     = (const float*)d_in[4];
    const float* b_e1     = (const float*)d_in[5];
    const float* W_e2     = (const float*)d_in[6];
    const float* b_e2     = (const float*)d_in[7];
    const float* W_d1     = (const float*)d_in[8];
    const float* b_d1     = (const float*)d_in[9];
    const float* W_d2     = (const float*)d_in[10];
    const float* b_d2     = (const float*)d_in[11];
    float* out = (float*)d_out;
    float* ws  = (float*)d_ws;

    float* xall = ws + WS_XALL;   // [640,4096]
    float* h    = ws + WS_H;      // [640,2048]
    float* enc  = ws + WS_ENC;    // [640,1024]
    float* z    = ws + WS_Z;      // [128,512]
    float* h2   = ws + WS_H2;     // [128,2048]

    // 1) permuted/concatenated encoder input (zero-padded to 640 rows)
    pack_xall<<<(MPAD * CWDIM) / 256, 256, 0, stream>>>(x, pseudo, xall);
    // 2) h = relu(xall @ W_e1 + b_e1)    [640,4096]x[4096,2048]
    gemm_bf16<1><<<dim3(HENC / 128, MPAD / 128), 256, 0, stream>>>(xall, W_e1, b_e1, h, HENC, CWDIM);
    // 3) enc = h @ W_e2 + b_e2           [640,2048]x[2048,1024]
    gemm_bf16<0><<<dim3((2 * ZD) / 128, MPAD / 128), 256, 0, stream>>>(h, W_e2, b_e2, enc, 2 * ZD, HENC);
    // 4) split + reparameterize
    reparam_kernel<<<(MROWS * ZD + 255) / 256, 256, 0, stream>>>(enc, eps, out, z);
    // 5) h2 = relu(z @ W_d1 + b_d1)      [128,512]x[512,2048]
    gemm_bf16<1><<<dim3(HDEC / 128, 1), 256, 0, stream>>>(z, W_d1, b_d1, h2, HDEC, ZD);
    // 6) cw_recon = h2 @ W_d2 + b_d2     [128,2048]x[2048,4096] -> d_out
    gemm_bf16<0><<<dim3(CWDIM / 128, 1), 256, 0, stream>>>(h2, W_d2, b_d2, out + OFF_CW, CWDIM, HDEC);
    // 7) cw_dist                          256 codes x 4 chunks of 128
    dist_kernel<<<dim3(BOOK / 128, DIMC), 256, 0, stream>>>(out + OFF_CW, codebook, out + OFF_DIST);
    // 8) argmin over book, idx as float
    argmin_kernel<<<(BATCH * DIMC) / 8, 256, 0, stream>>>(out + OFF_DIST, out + OFF_IDX);
}
